// ClusterBlock_79740362817810
// MI455X (gfx1250) — matrix operand order repaired and checked by reading, not compiled, not run on hardware
//
#include <hip/hip_runtime.h>
#include <hip/hip_bf16.h>
#include <cstdint>

typedef __attribute__((ext_vector_type(16))) _Float16 v16h;
typedef __attribute__((ext_vector_type(8)))  _Float16 v8h;
typedef __attribute__((ext_vector_type(4)))  _Float16 v4h;
typedef __attribute__((ext_vector_type(8)))  float    v8f;

#define SLOPE   325.0f
#define NEG_INF -9.0e15f

// Problem dims (fixed by reference setup_inputs)
constexpr int B_ = 16, V_ = 325, T_ = 12, F0_ = 64, F1_ = 64, K_ = 4, G_ = 3;
constexpr int MH   = B_ * V_ * T_;      // 62400 rows of the h-GEMM
constexpr int ND   = B_ * T_ * F1_;     // 12288 = hp-GEMM N dimension
constexpr int J0   = 163, J1 = 162;     // even/odd slice lengths of V
constexpr int JPAD = 192;               // K-padding for hp-GEMM (6 x 32)
constexpr int IPAD = 336;               // M-padding (21 x 16)
constexpr int MT_I = 21;                // i tiles
constexpr int NG   = ND / 64;           // 192 n-groups (each = one (b,t), 64 f)
constexpr int GC_SIZE = B_ * V_ * T_ * F1_;   // gc_act elements (3,993,600)

// Workspace layout (bytes); ~33.7 MB total
constexpr size_t XH_OFF  = 0;
constexpr size_t XH_SZ   = (size_t)MH * F0_ * 2;           // x in f16
constexpr size_t H_OFF   = XH_OFF + XH_SZ;
constexpr size_t H_SZ    = (size_t)2 * V_ * ND * 2;        // h[off][v][b,t,f] f16
constexpr size_t HT_OFF  = H_OFF + H_SZ;
constexpr size_t HT_SZ   = (size_t)2 * ND * JPAD * 2;      // hT[off][n][j] f16
constexpr size_t ATT_OFF = HT_OFF + HT_SZ;
constexpr size_t ATT_SZ  = (size_t)2 * IPAD * JPAD * 2;    // att f16 (zero-padded cols)
constexpr size_t S1_OFF  = ATT_OFF + ATT_SZ;
constexpr size_t S1_SZ   = (size_t)2 * JPAD * 4;
constexpr size_t S2_OFF  = S1_OFF + S1_SZ;

__device__ __forceinline__ float leaky(float x) { return x >= 0.f ? x : SLOPE * x; }

__device__ __forceinline__ v16h cat8(v8h lo, v8h hi) {
  return __builtin_shufflevector(lo, hi, 0,1,2,3,4,5,6,7,8,9,10,11,12,13,14,15);
}

// ---------------------------------------------------------------------------
// Kernel C: cluster = softmax_k( xv . cluster_W ), written straight to d_out.
// ---------------------------------------------------------------------------
__global__ void k_cluster(const float* __restrict__ x, const float* __restrict__ cw,
                          float* __restrict__ out_cl) {
  int p = blockIdx.x * blockDim.x + threadIdx.x;
  if (p >= B_ * V_) return;
  int b = p / V_, v = p % V_;
  const float* xv = x + (size_t)p * (T_ * F0_);
  float acc[G_ * K_];
#pragma unroll
  for (int i = 0; i < G_ * K_; i++) acc[i] = 0.f;
  for (int d = 0; d < T_ * F0_; d++) {
    float xd = xv[d];
#pragma unroll
    for (int i = 0; i < G_ * K_; i++) acc[i] += xd * cw[(size_t)i * (T_ * F0_) + d];
  }
  for (int g = 0; g < G_; g++) {
    float m = acc[g * 4];
#pragma unroll
    for (int k = 1; k < 4; k++) m = fmaxf(m, acc[g * 4 + k]);
    float e[4], s = 0.f;
#pragma unroll
    for (int k = 0; k < 4; k++) { e[k] = __expf(acc[g * 4 + k] - m); s += e[k]; }
    float inv = 1.f / s;
#pragma unroll
    for (int k = 0; k < 4; k++)
      out_cl[(((size_t)g * B_ + b) * V_ + v) * K_ + k] = e[k] * inv;
  }
}

// ---------------------------------------------------------------------------
// Kernel X: x (f32) -> xh (f16), 4 elements/thread.
// ---------------------------------------------------------------------------
__global__ void k_x2h(const float* __restrict__ x, _Float16* __restrict__ xh) {
  int i = blockIdx.x * 256 + threadIdx.x;            // exact: MH*F0/4 threads
  float4 v = reinterpret_cast<const float4*>(x)[i];
  v4h o;
  o[0] = (_Float16)v.x; o[1] = (_Float16)v.y; o[2] = (_Float16)v.z; o[3] = (_Float16)v.w;
  reinterpret_cast<v4h*>(xh)[i] = o;
}

// ---------------------------------------------------------------------------
// Kernel H: h = leaky(x . W^T + Wb) for both offsets of one (g,k).
// WMMA GEMM: M=62400, N=64, K=64. Writes h[v][n] and parity-sliced hT[n][j].
// Row decode (divisions by 12/325) hoisted out of the nt loop.
// ---------------------------------------------------------------------------
__global__ __launch_bounds__(256) void k_h(const _Float16* __restrict__ xh,
                                           const float* __restrict__ W,
                                           const float* __restrict__ Wb,
                                           _Float16* __restrict__ h_ws,
                                           _Float16* __restrict__ hT_ws,
                                           int gk) {
  // LDS holds B = W^T (K=c, N=f) in WMMA pair layout: pair p=c>>1 at u32 [p*64+f]
  __shared__ uint32_t ldsW[2048];
  int off = blockIdx.y;
  int tid = threadIdx.x;
  const float* Wp = W + ((size_t)gk * 2 + off) * (F1_ * F0_);
  for (int idx = tid; idx < F1_ * F0_; idx += 256) {
    int f = idx >> 6, c = idx & 63;
    reinterpret_cast<_Float16*>(ldsW)[(c >> 1) * 128 + f * 2 + (c & 1)] =
        (_Float16)Wp[(size_t)f * F0_ + c];
  }
  __syncthreads();

  int wave = tid >> 5, lane = tid & 31;
  int m_tile = blockIdx.x * 8 + wave;
  if (m_tile >= MH / 16) return;
  int col = lane & 15, hi = lane >> 4;
  const float* Wbp = Wb + ((size_t)gk * 2 + off) * F1_;

  // Per-row output addressing, computed once per wave.
  size_t hbase[8];
  size_t hTbase[8];
  bool   par[8];
#pragma unroll
  for (int r = 0; r < 8; r++) {
    int m  = m_tile * 16 + hi * 8 + r;           // C layout: rows r / 8+r
    int t  = m % T_;
    int v  = (m / T_) % V_;
    int bb = m / (T_ * V_);
    int nb = (bb * T_ + t) * F1_;
    hbase[r]  = ((size_t)off * V_ + v) * ND + nb;
    par[r]    = ((v & 1) == off);
    hTbase[r] = ((size_t)off * ND + nb) * JPAD + (v >> 1);
  }

#pragma unroll
  for (int nt = 0; nt < 4; nt++) {
    v8f acc = {};
#pragma unroll
    for (int s = 0; s < 2; s++) {
      // A tile 16x32: lane row = col, K halves split by hi
      const _Float16* ap = xh + (size_t)(m_tile * 16 + col) * F0_ + 32 * s + 8 * hi;
      v16h A = cat8(*reinterpret_cast<const v8h*>(ap),
                    *reinterpret_cast<const v8h*>(ap + 16));
      // B tile 32x16 from LDS: VGPR r holds K pair (2p, 2p+1) at column f
      int f = nt * 16 + col;
      v16h Bv;
#pragma unroll
      for (int r = 0; r < 8; r++) {
        union { uint32_t u; _Float16 h[2]; } cv;
        cv.u = ldsW[(16 * s + 4 * hi + r + (r >= 4 ? 4 : 0)) * 64 + f];
        Bv[2 * r] = cv.h[0]; Bv[2 * r + 1] = cv.h[1];
      }
      acc = __builtin_amdgcn_wmma_f32_16x16x32_f16(false, A, false, Bv,
                                                   (short)0, acc, false, false);
    }
    int f = nt * 16 + col;
    float bias = Wbp[f];
#pragma unroll
    for (int r = 0; r < 8; r++) {
      _Float16 h16 = (_Float16)leaky(acc[r] + bias);
      h_ws[hbase[r] + f] = h16;
      if (par[r]) hT_ws[hTbase[r] + (size_t)f * JPAD] = h16;
    }
  }
}

// ---------------------------------------------------------------------------
// Kernel T1: ht[v] = (sum_{b,t} h * Wt[t]) / B ; then s1/s2 dot products.
// ---------------------------------------------------------------------------
__global__ __launch_bounds__(64) void k_ht(const _Float16* __restrict__ h_ws,
                                           const float* __restrict__ Wt,
                                           const float* __restrict__ a,
                                           float* __restrict__ s1,
                                           float* __restrict__ s2, int gk) {
  int v = blockIdx.x, off = blockIdx.y, f = threadIdx.x;
  const float* Wtp = Wt + ((size_t)gk * 2 + off) * T_;
  float wt[T_];
#pragma unroll
  for (int t = 0; t < T_; t++) wt[t] = Wtp[t];
  const _Float16* hp = h_ws + ((size_t)off * V_ + v) * ND;
  float acc = 0.f;
  for (int bb = 0; bb < B_; bb++) {
    const _Float16* hb = hp + bb * T_ * F1_;
#pragma unroll
    for (int t = 0; t < T_; t++) acc += (float)hb[t * F1_ + f] * wt[t];
  }
  float ht = acc * (1.0f / B_);
  const float* ap = a + ((size_t)gk * 2 + off) * (2 * F1_);
  __shared__ float r1[64], r2[64];
  r1[f] = ht * ap[f];            // a1
  r2[f] = ht * ap[F1_ + f];      // a2
  __syncthreads();
  if (f == 0) {
    float x1 = 0.f, x2 = 0.f;
    for (int i = 0; i < 64; i++) { x1 += r1[i]; x2 += r2[i]; }
    s2[off * IPAD + v] = x2;
    if ((v & 1) == off) s1[off * JPAD + (v >> 1)] = x1;
  }
}

// ---------------------------------------------------------------------------
// Kernel T2: masked softmax attention row (one block per (i, off)).
// Writes f16 att with explicit zeros in padded K columns.
// ---------------------------------------------------------------------------
__global__ __launch_bounds__(256) void k_att(const int* __restrict__ adj,
                                             const float* __restrict__ s1,
                                             const float* __restrict__ s2,
                                             _Float16* __restrict__ att, int g) {
  int i = blockIdx.x, off = blockIdx.y, j = threadIdx.x;
  int J = off ? J1 : J0;
  __shared__ float red[256];
  float e = NEG_INF;
  if (j < J) {
    float val = leaky(s1[off * JPAD + j] + s2[off * IPAD + i]);
    int vj = off + 2 * j;
    e = (adj[((size_t)g * V_ + i) * V_ + vj] > 0) ? val : NEG_INF;
  }
  red[j] = e;
  __syncthreads();
  for (int s = 128; s > 0; s >>= 1) {
    if (j < s) red[j] = fmaxf(red[j], red[j + s]);
    __syncthreads();
  }
  float mx = red[0];
  __syncthreads();
  float ex = (j < J) ? __expf(e - mx) : 0.f;
  red[j] = ex;
  __syncthreads();
  for (int s = 128; s > 0; s >>= 1) {
    if (j < s) red[j] += red[j + s];
    __syncthreads();
  }
  float inv = 1.f / red[0];
  if (j < JPAD)
    att[((size_t)off * IPAD + i) * JPAD + j] = (j < J) ? (_Float16)(ex * inv)
                                                       : (_Float16)0.f;
}

// ---------------------------------------------------------------------------
// Kernel P: hp = att . hT for both offsets. One block = one 64-wide n-group
// (a single (b,t) pair); its hT panel (2 off x 64 n x 192 K f16 = 48 KB) is
// staged in LDS once and shared by 8 waves covering 8 i-tiles. Each wave
// reuses one A fragment across 4 n-tiles -> 48 WMMAs per wave. Epilogue:
// out += cluster/G * (leaky(hp0) + leaky(hp1)); exclusive tile ownership ->
// deterministic accumulation across the 12 sequential gk launches.
// ---------------------------------------------------------------------------
__global__ __launch_bounds__(256) void k_hp(const _Float16* __restrict__ att,
                                            const _Float16* __restrict__ hT,
                                            const float* __restrict__ cluster,
                                            float* __restrict__ out,
                                            int gk, int beta) {
  __shared__ _Float16 ldsB[2 * 64 * JPAD];           // 48 KB
  int n_group = blockIdx.x / 3;                      // 192 n-groups
  int mblk    = blockIdx.x % 3;                      // 3 m-blocks of 8 i-tiles
  int tid = threadIdx.x;

  {  // cooperative stage of both offsets' hT panels (two contiguous 24 KB runs)
    const uint4* s0 = reinterpret_cast<const uint4*>(hT + (size_t)(n_group * 64) * JPAD);
    const uint4* s1p = reinterpret_cast<const uint4*>(hT + ((size_t)ND + n_group * 64) * JPAD);
    uint4* dst = reinterpret_cast<uint4*>(ldsB);
    for (int i = tid; i < 1536; i += 256) {
      dst[i]        = s0[i];
      dst[1536 + i] = s1p[i];
    }
  }
  __syncthreads();

  int wave = tid >> 5, lane = tid & 31;
  int m_tile = mblk * 8 + wave;
  if (m_tile >= MT_I) return;                        // 24 waves cover 21 tiles
  int col = lane & 15, hi = lane >> 4;
  int g = gk >> 2, k = gk & 3;

  v8f acc[2][4] = {};
#pragma unroll
  for (int off = 0; off < 2; off++) {
    const _Float16* arow = att + ((size_t)off * IPAD + m_tile * 16 + col) * JPAD;
#pragma unroll
    for (int s = 0; s < 6; s++) {
      const _Float16* ap = arow + 32 * s + 8 * hi;
      __builtin_prefetch(arow + 32 * (s + 1), 0, 1);
      v16h A = cat8(*reinterpret_cast<const v8h*>(ap),
                    *reinterpret_cast<const v8h*>(ap + 16));
#pragma unroll
      for (int q = 0; q < 4; q++) {
        const _Float16* bp = &ldsB[(off * 64 + q * 16 + col) * JPAD + 32 * s + 8 * hi];
        v16h Bv = cat8(*reinterpret_cast<const v8h*>(bp),
                       *reinterpret_cast<const v8h*>(bp + 16));
        acc[off][q] = __builtin_amdgcn_wmma_f32_16x16x32_f16(false, A, false, Bv,
                                                             (short)0, acc[off][q],
                                                             false, false);
      }
    }
  }

  int bb = n_group / T_, t = n_group % T_;           // shared by all 4 q-tiles
#pragma unroll
  for (int r = 0; r < 8; r++) {
    int i = m_tile * 16 + hi * 8 + r;
    if (i < V_) {
      float cl = cluster[(((size_t)g * B_ + bb) * V_ + i) * K_ + k] * (1.0f / G_);
      size_t ob = (((size_t)bb * V_ + i) * T_ + t) * F1_ + col;
#pragma unroll
      for (int q = 0; q < 4; q++) {
        float val = (leaky(acc[0][q][r]) + leaky(acc[1][q][r])) * cl;
        size_t oi = ob + q * 16;
        out[oi] = beta ? (out[oi] + val) : val;
      }
    }
  }
}

// ---------------------------------------------------------------------------
extern "C" void kernel_launch(void* const* d_in, const int* in_sizes, int n_in,
                              void* d_out, int out_size, void* d_ws, size_t ws_size,
                              hipStream_t stream) {
  (void)in_sizes; (void)n_in; (void)out_size; (void)ws_size;
  const float* x   = (const float*)d_in[0];
  const int*   adj = (const int*)d_in[1];
  const float* W   = (const float*)d_in[2];
  const float* Wb  = (const float*)d_in[3];
  const float* Wt  = (const float*)d_in[4];
  const float* a   = (const float*)d_in[5];
  const float* cw  = (const float*)d_in[6];
  float* out = (float*)d_out;
  char*  ws  = (char*)d_ws;

  _Float16* xh  = (_Float16*)(ws + XH_OFF);
  _Float16* hws = (_Float16*)(ws + H_OFF);
  _Float16* hT  = (_Float16*)(ws + HT_OFF);
  _Float16* att = (_Float16*)(ws + ATT_OFF);
  float* s1 = (float*)(ws + S1_OFF);
  float* s2 = (float*)(ws + S2_OFF);
  float* cl = out + GC_SIZE;               // cluster output region of d_out

  k_cluster<<<(B_ * V_ + 255) / 256, 256, 0, stream>>>(x, cw, cl);
  k_x2h<<<(MH * F0_ / 4) / 256, 256, 0, stream>>>(x, xh);

  for (int gk = 0; gk < G_ * K_; gk++) {
    k_h  <<<dim3(488, 2), 256, 0, stream>>>(xh, W, Wb, hws, hT, gk);
    k_ht <<<dim3(V_, 2),   64, 0, stream>>>(hws, Wt, a, s1, s2, gk);
    k_att<<<dim3(V_, 2),  256, 0, stream>>>(adj, s1, s2, att, gk >> 2);
    k_hp <<<NG * 3, 256, 0, stream>>>(att, hT, cl, out, gk, gk > 0);
  }
}
